// FullyConnectedSteerableGeometricProductLayer_70196945486266
// MI455X (gfx1250) — matrix operand order repaired and checked by reading, not compiled, not run on hardware
//
#include <hip/hip_runtime.h>
#include <math.h>

// Problem constants (from reference setup_inputs)
#define BATCH  16
#define SEQ    618
#define NBLADE 8
#define NSUB   4
#define NTILE  ((SEQ + 15) / 16)   // 39 column tiles
#define KFULL  ((SEQ / 32) * 32)   // 608: last full 32-chunk boundary

typedef __attribute__((ext_vector_type(16))) _Float16 v16h;
typedef __attribute__((ext_vector_type(8)))  float    v8f;
typedef __attribute__((ext_vector_type(4)))  float    v4f;

// ---------------------------------------------------------------------------
// Kernel 1: q/k = normalize(mv_linear(x, w, b), a)
//   GEMM per blade component i:  out[b, n, i] = sum_m x[b,m,i] * w[n,m,sub(i)]
//   M = batch (16) -> one WMMA tile row; N = n (39 tiles); K = m (19 full
//   32-chunks + 1 guarded tail chunk).
// One block = 8 waves; wave w handles blade i = w. Epilogue: bias -> LDS ->
// grade-wise normalization -> workspace (f32).
// ---------------------------------------------------------------------------
__global__ __launch_bounds__(256)
void qk_linear_normalize_kernel(const float* __restrict__ x,
                                const float* __restrict__ qw,
                                const float* __restrict__ qb,
                                const float* __restrict__ kw,
                                const float* __restrict__ kb,
                                const float* __restrict__ a,
                                float* __restrict__ qout,
                                float* __restrict__ kout)
{
    __shared__ float tile[BATCH * 16 * NBLADE];   // [b][n_local][blade], 8 KB

    const int tid  = threadIdx.x;
    const int i    = tid >> 5;          // blade component (wave id, 0..7)
    const int lane = tid & 31;
    const int g    = lane >> 4;         // lane half-group (0 or 1)
    const int lr   = lane & 15;

    // blade -> grade subspace: [0,1,1,1,2,2,2,3]
    const int s = (i == 0) ? 0 : (i <= 3) ? 1 : (i <= 6) ? 2 : 3;

    const int  n0     = blockIdx.x * 16;
    const bool is_k   = (blockIdx.y != 0);
    const float* w    = is_k ? kw : qw;
    const float* bias = is_k ? kb : qb;
    float* outbuf     = is_k ? kout : qout;

    const int n_col = n0 + lr;                            // B/D column
    const int n_ld  = (n_col < SEQ) ? n_col : (SEQ - 1);  // clamped: garbage
    // only pollutes D columns n>=SEQ, which are never stored.

    // Per-lane loop-invariant base pointers.
    const float* __restrict__ xbase = x + (size_t)lr * SEQ * NBLADE + i;
    const float* __restrict__ wbase = w + (size_t)n_ld * SEQ * NSUB + s;

    // Fragment K maps (ISA 7.12.2, wave32):
    //   A (16x32, M=lane%16): VGPR v, half h -> K = v<4 ? g*8+2v+h : 16+g*8+2(v-4)+h
    //   B (32x16, N=lane%16): VGPR v, half h -> K = same as A
    v8f acc = {};

    // ---- 19 full K-chunks: no guards -> clauseable loads, one wait ----
    for (int m0 = 0; m0 < KFULL; m0 += 32) {
        v16h af, bf;
        #pragma unroll
        for (int v = 0; v < 8; ++v) {
            #pragma unroll
            for (int h = 0; h < 2; ++h) {
                const int Ka = (v < 4) ? (g * 8 + v * 2 + h)
                                       : (16 + g * 8 + (v - 4) * 2 + h);
                const int Kb = Ka;
                af[v * 2 + h] = (_Float16)xbase[(size_t)(m0 + Ka) * NBLADE];
                bf[v * 2 + h] = (_Float16)wbase[(size_t)(m0 + Kb) * NSUB];
            }
        }
        acc = __builtin_amdgcn_wmma_f32_16x16x32_f16(
                  false, af, false, bf, (short)0, acc, false, false);
    }

    // ---- tail chunk (m = 608..617 valid): clamp address, select value ----
    {
        const int m0 = KFULL;
        v16h af, bf;
        #pragma unroll
        for (int v = 0; v < 8; ++v) {
            #pragma unroll
            for (int h = 0; h < 2; ++h) {
                const int Ka = (v < 4) ? (g * 8 + v * 2 + h)
                                       : (16 + g * 8 + (v - 4) * 2 + h);
                const int Kb = Ka;
                const int ma = m0 + Ka, mb = m0 + Kb;
                const int mac = (ma < SEQ) ? ma : (SEQ - 1);
                const int mbc = (mb < SEQ) ? mb : (SEQ - 1);
                float va = xbase[(size_t)mac * NBLADE];   // always legal
                float vb = wbase[(size_t)mbc * NSUB];
                va = (ma < SEQ) ? va : 0.0f;              // v_cndmask, no branch
                vb = (mb < SEQ) ? vb : 0.0f;
                af[v * 2 + h] = (_Float16)va;
                bf[v * 2 + h] = (_Float16)vb;
            }
        }
        acc = __builtin_amdgcn_wmma_f32_16x16x32_f16(
                  false, af, false, bf, (short)0, acc, false, false);
    }

    // Bias goes only into the scalar blade (i == 0)
    const float bval = (i == 0) ? bias[n_ld] : 0.0f;   // discarded if n>=SEQ
    // D layout: VGPR r, lanes 0-15 -> M=r, lanes 16-31 -> M=r+8; N=lane%16
    #pragma unroll
    for (int r = 0; r < 8; ++r) {
        const int bb = r + 8 * g;       // batch row
        tile[(bb * 16 + lr) * NBLADE + i] = acc[r] + bval;
    }
    __syncthreads();

    // Grade-wise normalization: one thread per (b, n_local) pair (16x16=256)
    const int b  = tid >> 4;
    const int nl = tid & 15;
    const int n  = n0 + nl;
    if (n < SEQ) {
        float v[8];
        #pragma unroll
        for (int j = 0; j < 8; ++j) v[j] = tile[(b * 16 + nl) * NBLADE + j];

        float gn0 = sqrtf(v[0] * v[0]);
        float gn1 = sqrtf(v[1] * v[1] + v[2] * v[2] + v[3] * v[3]);
        float gn2 = sqrtf(v[4] * v[4] + v[5] * v[5] + v[6] * v[6]);
        float gn3 = sqrtf(v[7] * v[7]);

        const float sa0 = 1.0f / (1.0f + expf(-a[n * NSUB + 0]));
        const float sa1 = 1.0f / (1.0f + expf(-a[n * NSUB + 1]));
        const float sa2 = 1.0f / (1.0f + expf(-a[n * NSUB + 2]));
        const float sa3 = 1.0f / (1.0f + expf(-a[n * NSUB + 3]));
        gn0 = sa0 * (gn0 - 1.0f) + 1.0f + 1e-6f;
        gn1 = sa1 * (gn1 - 1.0f) + 1.0f + 1e-6f;
        gn2 = sa2 * (gn2 - 1.0f) + 1.0f + 1e-6f;
        gn3 = sa3 * (gn3 - 1.0f) + 1.0f + 1e-6f;

        const float r0 = 1.0f / gn0, r1 = 1.0f / gn1;
        const float r2 = 1.0f / gn2, r3 = 1.0f / gn3;
        v4f lo, hi;
        lo.x = v[0] * r0; lo.y = v[1] * r1; lo.z = v[2] * r1; lo.w = v[3] * r1;
        hi.x = v[4] * r2; hi.y = v[5] * r2; hi.z = v[6] * r2; hi.w = v[7] * r3;
        v4f* o = (v4f*)(outbuf + (size_t)(b * SEQ + n) * NBLADE);
        o[0] = lo;
        o[1] = hi;
    }
}

// ---------------------------------------------------------------------------
// Kernel 2: pairwise geometric product (Cl(3,0), Euclidean).
//   out[b,m,n,j] = sum_{i,k} q[b,m,i] C[i,j,k] k[b,n,k]   (64 nonzeros)
// Output-bandwidth bound (195.6 MB) -> 128-bit non-temporal stores.
// ---------------------------------------------------------------------------
__global__ __launch_bounds__(256)
void geo_product_kernel(const float* __restrict__ q,
                        const float* __restrict__ k,
                        float* __restrict__ out)
{
    const int n = blockIdx.x * blockDim.x + threadIdx.x;
    const int m = blockIdx.y;
    const int b = blockIdx.z;
    if (n >= SEQ) return;

    const v4f* qp = (const v4f*)(q + (size_t)(b * SEQ + m) * NBLADE);
    const v4f* kp = (const v4f*)(k + (size_t)(b * SEQ + n) * NBLADE);
    const v4f qlo = qp[0], qhi = qp[1];
    const v4f klo = kp[0], khi = kp[1];
    const float q0 = qlo.x, q1 = qlo.y, q2 = qlo.z, q3 = qlo.w;
    const float q4 = qhi.x, q5 = qhi.y, q6 = qhi.z, q7 = qhi.w;
    const float k0 = klo.x, k1 = klo.y, k2 = klo.z, k3 = klo.w;
    const float k4 = khi.x, k5 = khi.y, k6 = khi.z, k7 = khi.w;

    // Blade order [1, e1, e2, e3, e12, e13, e23, e123]; signs from the
    // reference's basis-swap-count rule (e12^2 = e13^2 = e23^2 = e123^2 = -1).
    v4f olo, ohi;
    olo.x = q0*k0 + q1*k1 + q2*k2 + q3*k3 - q4*k4 - q5*k5 - q6*k6 - q7*k7;
    olo.y = q0*k1 + q1*k0 - q2*k4 - q3*k5 + q4*k2 + q5*k3 - q6*k7 - q7*k6;
    olo.z = q0*k2 + q1*k4 + q2*k0 - q3*k6 - q4*k1 + q5*k7 + q6*k3 + q7*k5;
    olo.w = q0*k3 + q1*k5 + q2*k6 + q3*k0 - q4*k7 - q5*k1 - q6*k2 - q7*k4;
    ohi.x = q0*k4 + q1*k2 - q2*k1 + q3*k7 + q4*k0 - q5*k6 + q6*k5 + q7*k3;
    ohi.y = q0*k5 + q1*k3 - q2*k7 - q3*k1 + q4*k6 + q5*k0 - q6*k4 - q7*k2;
    ohi.z = q0*k6 + q1*k7 + q2*k3 - q3*k2 - q4*k5 + q5*k4 + q6*k0 + q7*k1;
    ohi.w = q0*k7 + q1*k6 - q2*k5 + q3*k4 + q4*k3 - q5*k2 + q6*k1 + q7*k0;

    v4f* dst = (v4f*)(out + (((size_t)(b * SEQ + m) * SEQ) + n) * NBLADE);
    __builtin_nontemporal_store(olo, dst);       // never re-read: bypass RT
    __builtin_nontemporal_store(ohi, dst + 1);
}

// ---------------------------------------------------------------------------
extern "C" void kernel_launch(void* const* d_in, const int* in_sizes, int n_in,
                              void* d_out, int out_size, void* d_ws, size_t ws_size,
                              hipStream_t stream)
{
    const float* x  = (const float*)d_in[0];   // [16, 618, 8]
    const float* qw = (const float*)d_in[1];   // [618, 618, 4]
    const float* qb = (const float*)d_in[2];   // [1, 618, 1]
    const float* kw = (const float*)d_in[3];   // [618, 618, 4]
    const float* kb = (const float*)d_in[4];   // [1, 618, 1]
    const float* a  = (const float*)d_in[5];   // [618, 4]

    float* qbuf = (float*)d_ws;                           // 316 KB
    float* kbuf = qbuf + (size_t)BATCH * SEQ * NBLADE;    // 316 KB
    float* out  = (float*)d_out;                          // [16,618,618,8]

    qk_linear_normalize_kernel<<<dim3(NTILE, 2), 256, 0, stream>>>(
        x, qw, qb, kw, kb, a, qbuf, kbuf);

    geo_product_kernel<<<dim3((SEQ + 255) / 256, SEQ, BATCH), 256, 0, stream>>>(
        qbuf, kbuf, out);
}
